// GPT2Attention_4904852652167
// MI455X (gfx1250) — matrix operand order repaired and checked by reading, not compiled, not run on hardware
//
#include <hip/hip_runtime.h>
#include <hip/hip_bf16.h>
#include <math.h>
#include <stdint.h>

typedef __attribute__((ext_vector_type(16))) _Float16 v16h;
typedef __attribute__((ext_vector_type(8)))  _Float16 v8h;
typedef __attribute__((ext_vector_type(4)))  _Float16 v4h;
typedef __attribute__((ext_vector_type(8)))  float    v8f;
typedef __attribute__((ext_vector_type(4)))  float    f4;

#define LOG2E 1.4426950408889634f

__device__ __forceinline__ v16h cat16(v8h lo, v8h hi) {
  return __builtin_shufflevector(lo, hi, 0,1,2,3,4,5,6,7,8,9,10,11,12,13,14,15);
}

__device__ __forceinline__ v8f wmma16x16x32(v16h a, v16h b, v8f c) {
  return __builtin_amdgcn_wmma_f32_16x16x32_f16(false, a, false, b, (short)0, c, false, false);
}

// ---- CDNA5 async global->LDS copy (ASYNCcnt path), 16B per lane ----
__device__ __forceinline__ uint32_t lds_of(const void* p) {
  return (uint32_t)(uintptr_t)p;   // generic LDS pointer low 32 bits == LDS offset
}
__device__ __forceinline__ void async_b128(uint32_t lds_off, const void* gptr) {
  asm volatile("global_load_async_to_lds_b128 %0, %1, off"
               :: "v"(lds_off), "v"((uint64_t)(uintptr_t)gptr) : "memory");
}
__device__ __forceinline__ void wait_async0() {
  asm volatile("s_wait_asynccnt 0x0" ::: "memory");
}

// A fragment (16x32, f16) from row-major storage [row][ld].
__device__ __forceinline__ v16h load_a_frag(const _Float16* base, int ld, int row0, int k0, int lane) {
  const _Float16* p = base + (size_t)(row0 + (lane & 15)) * ld + k0 + ((lane >> 4) * 8);
  return cat16(*(const v8h*)p, *(const v8h*)(p + 16));
}

// B fragment (32x16, f16) from K-contiguous storage bT[n][ld].
__device__ __forceinline__ v16h load_b_frag(const _Float16* base, int ld, int n0, int k0, int lane) {
  const _Float16* p = base + (size_t)(n0 + (lane & 15)) * ld + k0 + ((lane >> 4) * 8);
  return cat16(*(const v8h*)p, *(const v8h*)(p + 16));
}

__global__ void cvt_f32_to_f16(const float* __restrict__ in, _Float16* __restrict__ out, int n4) {
  int i = blockIdx.x * blockDim.x + threadIdx.x;
  if (i < n4) {
    f4 v = ((const f4*)in)[i];
    v4h o;
#pragma unroll
    for (int j = 0; j < 4; ++j) o[j] = (_Float16)v[j];
    ((v4h*)out)[i] = o;
  }
}

// Transpose-convert weights once: in f32 [K][N] -> out f16 [N][K]. 32x32 LDS tiles.
__global__ __launch_bounds__(256)
void cvt_transpose_f16(const float* __restrict__ in, _Float16* __restrict__ out, int K, int N) {
  __shared__ float t[32][33];
  const int k0 = blockIdx.y * 32, n0 = blockIdx.x * 32;
  const int tx = threadIdx.x & 31, ty = threadIdx.x >> 5;
#pragma unroll
  for (int i = 0; i < 4; ++i) t[ty + i * 8][tx] = in[(size_t)(k0 + ty + i * 8) * N + n0 + tx];
  __syncthreads();
#pragma unroll
  for (int i = 0; i < 4; ++i)
    out[(size_t)(n0 + ty + i * 8) * K + k0 + tx] = (_Float16)t[tx][ty + i * 8];
}

// C = A[MxK] * BT[NxK]^T + bias. Block tile 128x256, BK=64, wave tile 64x64.
// Double-buffered async LDS staging.
// OUT_MODE 0: f32 out [M][N].  OUT_MODE 1: QKV split -> qk f16 [M][4096], V -> vT f16 [B*H*128][S].
template<int OUT_MODE>
__global__ __launch_bounds__(256, 1)
void gemm_wmma(const _Float16* __restrict__ A, const _Float16* __restrict__ BT,
               const float* __restrict__ bias, void* __restrict__ out0, void* __restrict__ out1,
               int M, int N, int K, int S) {
  constexpr int BM = 128, BN = 256, BK = 64;
  __shared__ __align__(16) _Float16 lA[2][BM][BK];   // 32 KB
  __shared__ __align__(16) _Float16 lBT[2][BN][BK];  // 64 KB
  const int tid  = threadIdx.x;
  const int lane = tid & 31;
  const int wave = tid >> 5;
  const int wm = (wave & 1) * 64;    // 2 waves along M
  const int wn = (wave >> 1) * 64;   // 4 waves along N
  const int blockM = blockIdx.y * BM;
  const int blockN = blockIdx.x * BN;

  v8f acc[4][4] = {};

  auto issue = [&](int k0, int buf) {
#pragma unroll
    for (int p = 0; p < 4; ++p) {    // A: 128x64 halves = 1024 b128 chunks
      int idx = p * 256 + tid;
      int r = idx >> 3, c = (idx & 7) * 8;
      async_b128(lds_of(&lA[buf][r][c]), A + (size_t)(blockM + r) * K + k0 + c);
    }
#pragma unroll
    for (int p = 0; p < 8; ++p) {    // BT: 256x64 halves = 2048 b128 chunks
      int idx = p * 256 + tid;
      int r = idx >> 3, c = (idx & 7) * 8;
      async_b128(lds_of(&lBT[buf][r][c]), BT + (size_t)(blockN + r) * K + k0 + c);
    }
  };

  const int nIter = K / BK;
  issue(0, 0);
  for (int it = 0; it < nIter; ++it) {
    wait_async0();
    __syncthreads();
    if (it + 1 < nIter) issue((it + 1) * BK, (it + 1) & 1);
    const _Float16* a  = &lA[it & 1][0][0];
    const _Float16* bt = &lBT[it & 1][0][0];
#pragma unroll
    for (int ks = 0; ks < BK; ks += 32) {
      v16h af[4], bf[4];
#pragma unroll
      for (int mt = 0; mt < 4; ++mt) af[mt] = load_a_frag(a, BK, wm + mt * 16, ks, lane);
#pragma unroll
      for (int nt = 0; nt < 4; ++nt) bf[nt] = load_b_frag(bt, BK, wn + nt * 16, ks, lane);
#pragma unroll
      for (int mt = 0; mt < 4; ++mt)
#pragma unroll
        for (int nt = 0; nt < 4; ++nt)
          acc[mt][nt] = wmma16x16x32(af[mt], bf[nt], acc[mt][nt]);
    }
  }

#pragma unroll
  for (int mt = 0; mt < 4; ++mt) {
#pragma unroll
    for (int nt = 0; nt < 4; ++nt) {
      const int nb = blockN + wn + nt * 16;          // tile base col (uniform)
      const int n  = nb + (lane & 15);
      const int m0 = blockM + wm + mt * 16 + ((lane >> 4) * 8);
      const float bv = bias[n];
      if (OUT_MODE == 0) {
        float* C = (float*)out0;
#pragma unroll
        for (int r = 0; r < 8; ++r) C[(size_t)(m0 + r) * N + n] = acc[mt][nt][r] + bv;
      } else {
        if (nb < 4096) {  // Q,K -> [M][4096] f16
          _Float16* qkp = (_Float16*)out0;
#pragma unroll
          for (int r = 0; r < 8; ++r)
            qkp[(size_t)(m0 + r) * 4096 + n] = (_Float16)(acc[mt][nt][r] + bv);
        } else {          // V -> vT[(b*16+h)*128 + d][s], rows r are consecutive s
          const int c = n - 4096;
          const int h = c >> 7, d = c & 127;
          const int b = m0 >> 11, s0 = m0 & 2047;    // S == 2048
          v8h hv;
#pragma unroll
          for (int r = 0; r < 8; ++r) hv[r] = (_Float16)(acc[mt][nt][r] + bv);
          _Float16* vt = (_Float16*)out1;
          *(v8h*)(vt + ((size_t)(b * 16 + h) * 128 + d) * S + s0) = hv;
        }
      }
    }
  }
}

// Causal flash attention. qk f16 [B,S,4096]; vT f16 [B*H*128][S]; out f16 [B,S,2048].
// Grid (S/128, H, B); 8 waves x 16 query rows; 64-key tiles double-buffered via async copies.
__global__ __launch_bounds__(256, 1)
void attn_wmma(const _Float16* __restrict__ qk, const _Float16* __restrict__ vT,
               _Float16* __restrict__ outh, int S) {
  constexpr int DM = 2048, LDQK = 4096, DH = 128, KT = 64;
  __shared__ __align__(16) _Float16 lK[2][KT][DH];    // key-major      (32 KB)
  __shared__ __align__(16) _Float16 lVT[2][DH][KT];   // d-major        (32 KB)
  __shared__ __align__(16) _Float16 lP[8][16][KT];    // per-wave P     (16 KB)

  const int tid  = threadIdx.x;
  const int lane = tid & 31;
  const int wave = tid >> 5;
  const int qTile = blockIdx.x, h = blockIdx.y, b = blockIdx.z;
  const int qBase = qTile * 128;
  const size_t baseRow = (size_t)b * S;
  const int offQ = h * DH, offK = DM + h * DH;
  const size_t vRow0 = (size_t)(b * 16 + h) * 128;

  // Q fragments: 4 x (16x32) A-frags held in registers
  const int qRowFrag = qBase + wave * 16 + (lane & 15);
  v16h qf[4];
#pragma unroll
  for (int d = 0; d < 4; ++d) {
    const _Float16* p = qk + (baseRow + qRowFrag) * LDQK + offQ + d * 32 + ((lane >> 4) * 8);
    qf[d] = cat16(*(const v8h*)p, *(const v8h*)(p + 16));
  }

  v8f O[8] = {};
  float mrow[8], lrow[8];
#pragma unroll
  for (int r = 0; r < 8; ++r) { mrow[r] = -__builtin_inff(); lrow[r] = 0.f; }

  const float scale = 0.08838834764831845f;            // 1/sqrt(128)
  const int myQrow0 = qBase + wave * 16 + ((lane >> 4) * 8);
  const int numTiles = (qTile + 1) * 2;                // 64-key tiles up to qBase+128

  auto issueKV = [&](int kBase, int buf) {
#pragma unroll
    for (int p = 0; p < 4; ++p) {
      int idx = p * 256 + tid;
      {  // K tile [64][128]: 1024 b128 chunks
        int kr = idx >> 4, c = (idx & 15) * 8;
        async_b128(lds_of(&lK[buf][kr][c]),
                   qk + (baseRow + kBase + kr) * LDQK + offK + c);
      }
      {  // V^T tile [128][64]: 1024 b128 chunks, rows contiguous in vT
        int d = idx >> 3, c = (idx & 7) * 8;
        async_b128(lds_of(&lVT[buf][d][c]), vT + (vRow0 + d) * S + kBase + c);
      }
    }
  };

  issueKV(0, 0);
  for (int kt = 0; kt < numTiles; ++kt) {
    const int kBase = kt * KT;
    const int buf = kt & 1;
    wait_async0();
    __syncthreads();
    if (kt + 1 < numTiles) issueKV((kt + 1) * KT, (kt + 1) & 1);

    // scores S = Q * K^T : four 16x16 tiles, 4 k-steps over Dh
    v8f sc[4] = {};
#pragma unroll
    for (int d = 0; d < 4; ++d) {
#pragma unroll
      for (int j = 0; j < 4; ++j) {
        v16h kb = load_b_frag(&lK[buf][0][0], DH, j * 16, d * 32, lane);
        sc[j] = wmma16x16x32(qf[d], kb, sc[j]);
      }
    }

    // online softmax: each row lives across 16 lanes of a half-wave
    float pa[4][8], corr[8];
#pragma unroll
    for (int r = 0; r < 8; ++r) {
      const int qr = myQrow0 + r;
      float av[4];
#pragma unroll
      for (int j = 0; j < 4; ++j) {
        const int kIdx = kBase + j * 16 + (lane & 15);
        av[j] = (kIdx <= qr) ? sc[j][r] * scale : -__builtin_inff();
      }
      float mx = fmaxf(fmaxf(av[0], av[1]), fmaxf(av[2], av[3]));
#pragma unroll
      for (int off = 8; off >= 1; off >>= 1) mx = fmaxf(mx, __shfl_xor(mx, off, 32));
      const float mnew = fmaxf(mrow[r], mx);
      float sum = 0.f;
#pragma unroll
      for (int j = 0; j < 4; ++j) {
        const float p = exp2f((av[j] - mnew) * LOG2E);
        pa[j][r] = p;
        sum += p;
      }
      const float cf = exp2f((mrow[r] - mnew) * LOG2E);
#pragma unroll
      for (int off = 8; off >= 1; off >>= 1) sum += __shfl_xor(sum, off, 32);
      lrow[r] = lrow[r] * cf + sum;
      mrow[r] = mnew;
      corr[r] = cf;
    }
#pragma unroll
    for (int dt = 0; dt < 8; ++dt)
#pragma unroll
      for (int r = 0; r < 8; ++r) O[dt][r] *= corr[r];

    // P: C-layout -> A-layout via per-wave LDS round-trip
    {
      const int mr = (lane >> 4) * 8;
      const int col = lane & 15;
#pragma unroll
      for (int r = 0; r < 8; ++r)
#pragma unroll
        for (int j = 0; j < 4; ++j)
          lP[wave][mr + r][j * 16 + col] = (_Float16)pa[j][r];
    }
    v16h pf0 = load_a_frag(&lP[wave][0][0], KT, 0, 0,  lane);
    v16h pf1 = load_a_frag(&lP[wave][0][0], KT, 0, 32, lane);

    // O += P(16x64) * V(64x128)
#pragma unroll
    for (int dt = 0; dt < 8; ++dt) {
      v16h vb0 = load_b_frag(&lVT[buf][0][0], KT, dt * 16, 0,  lane);
      v16h vb1 = load_b_frag(&lVT[buf][0][0], KT, dt * 16, 32, lane);
      O[dt] = wmma16x16x32(pf0, vb0, O[dt]);
      O[dt] = wmma16x16x32(pf1, vb1, O[dt]);
    }
  }

#pragma unroll
  for (int dt = 0; dt < 8; ++dt) {
    const int col = h * DH + dt * 16 + (lane & 15);
#pragma unroll
    for (int r = 0; r < 8; ++r) {
      const int row = myQrow0 + r;
      outh[(baseRow + row) * DM + col] = (_Float16)(O[dt][r] / lrow[r]);
    }
  }
}

extern "C" void kernel_launch(void* const* d_in, const int* in_sizes, int n_in,
                              void* d_out, int out_size, void* d_ws, size_t ws_size,
                              hipStream_t stream) {
  const float* x     = (const float*)d_in[0];
  const float* w_qkv = (const float*)d_in[1];
  const float* b_qkv = (const float*)d_in[2];
  const float* w_out = (const float*)d_in[3];
  const float* b_out = (const float*)d_in[4];
  float* out = (float*)d_out;

  const int B = 4, S = 2048, DM = 2048, H = 16;
  const size_t Nx   = (size_t)B * S * DM;        // 16,777,216
  const size_t Nwq  = (size_t)DM * 3 * DM;       // 12,582,912
  const size_t Nwo  = (size_t)DM * DM;           //  4,194,304
  const size_t Nqk  = (size_t)B * S * 2 * DM;    // 33,554,432  (Q,K [M][4096])
  const size_t Nv   = (size_t)B * S * DM;        // 16,777,216  (V^T [B*H*128][S])

  char* ws = (char*)d_ws;
  _Float16* xh    = (_Float16*)ws; ws += Nx  * 2;
  _Float16* wqT   = (_Float16*)ws; ws += Nwq * 2;   // [6144][2048]
  _Float16* woT   = (_Float16*)ws; ws += Nwo * 2;   // [2048][2048]
  _Float16* qkh   = (_Float16*)ws; ws += Nqk * 2;
  _Float16* vTh   = (_Float16*)ws; ws += Nv  * 2;
  _Float16* attnh = (_Float16*)ws;

  cvt_f32_to_f16<<<(int)((Nx / 4 + 255) / 256), 256, 0, stream>>>(x, xh, (int)(Nx / 4));
  cvt_transpose_f16<<<dim3(3 * DM / 32, DM / 32), 256, 0, stream>>>(w_qkv, wqT, DM, 3 * DM);
  cvt_transpose_f16<<<dim3(DM / 32, DM / 32), 256, 0, stream>>>(w_out, woT, DM, DM);

  // qkv projection: Q,K -> qkh, V -> vTh (pre-transposed for attention)
  gemm_wmma<1><<<dim3(3 * DM / 256, B * S / 128), 256, 0, stream>>>(
      xh, wqT, b_qkv, (void*)qkh, (void*)vTh, B * S, 3 * DM, DM, S);

  // causal flash attention
  attn_wmma<<<dim3(S / 128, H, B), 256, 0, stream>>>(qkh, vTh, attnh, S);

  // output projection -> f32
  gemm_wmma<0><<<dim3(DM / 256, B * S / 128), 256, 0, stream>>>(
      attnh, woT, b_out, (void*)out, nullptr, B * S, DM, DM, S);
}
